// ScaledDotProductAttention_34746285425440
// MI455X (gfx1250) — compile-verified, matrix operand order checked by reading, not run on hardware
//
#include <hip/hip_runtime.h>

#define AB  2
#define AH  16
#define AS  2048
#define ADK 64
#define ADV 64

typedef __attribute__((ext_vector_type(16))) _Float16 v16h;
typedef __attribute__((ext_vector_type(8)))  _Float16 v8h;
typedef __attribute__((ext_vector_type(8)))  float    v8f;

// Load a 16x32 f16 WMMA operand fragment (A layout; B uses the same per-lane
// pattern with "row" = output column) from LDS via two ds_load_b128.
__device__ __forceinline__ v16h load_frag16(const _Float16* base, int rowBase,
                                            int stride, int k0, int lane) {
  const _Float16* p = base + (rowBase + (lane & 15)) * stride + k0 + ((lane >> 4) << 3);
  v8h lo = *(const v8h*)(p);        // K = k0+kh .. +7    -> VGPR 0..3
  v8h hi = *(const v8h*)(p + 16);   // K = k0+16+kh .. +7 -> VGPR 4..7
  v16h r;
#pragma unroll
  for (int i = 0; i < 8; ++i) { r[i] = lo[i]; r[i + 8] = hi[i]; }
  return r;
}

// Max-reduce across the 16 lanes of each lane-half (wave32) using
// ds_swizzle_b32 with immediate xor patterns (no index VGPRs, no bpermute).
__device__ __forceinline__ float rowmax16(float x) {
  x = fmaxf(x, __int_as_float(__builtin_amdgcn_ds_swizzle(__float_as_int(x), 0x041F))); // xor 1
  x = fmaxf(x, __int_as_float(__builtin_amdgcn_ds_swizzle(__float_as_int(x), 0x081F))); // xor 2
  x = fmaxf(x, __int_as_float(__builtin_amdgcn_ds_swizzle(__float_as_int(x), 0x101F))); // xor 4
  x = fmaxf(x, __int_as_float(__builtin_amdgcn_ds_swizzle(__float_as_int(x), 0x201F))); // xor 8
  return x;
}

__global__ __launch_bounds__(128) void fa_fwd_f16wmma(
    const float* __restrict__ Qg, const float* __restrict__ Kg,
    const float* __restrict__ Vg, float* __restrict__ Og) {
  __shared__ __align__(16) _Float16 Qs[64 * 64];     // Q tile, row-major, pre-scaled
  __shared__ __align__(16) _Float16 Ks[32 * 64];     // K tile, row-major
  __shared__ __align__(16) _Float16 VTs[64 * 32];    // V tile, transposed (dv x kt)
  __shared__ __align__(16) _Float16 Ps[4 * 16 * 32]; // per-wave P scratch

  const int tid  = threadIdx.x;
  const int lane = tid & 31;
  const int wave = tid >> 5;
  const int lh   = lane >> 4;   // lane half: row offset +0 / +8 in C frags
  const int ln   = lane & 15;

  const int bh = blockIdx.y;            // fused (b,h): 0..31
  const int q0 = blockIdx.x * 64;
  const size_t base = (size_t)bh * AS * ADK;

  // ---- Stage Q tile to LDS as f16, folding in 1/sqrt(DK) * log2(e) so all
  //      softmax math runs in the exp2 domain (v_exp_f32 is a base-2 exp). ----
  {
    const float scale = 0.125f * 1.44269504088896340736f;
    const int row = tid >> 1;
    const int cb  = (tid & 1) * 32;
    const float* src = Qg + base + (size_t)(q0 + row) * ADK + cb;
    _Float16* dst = Qs + row * 64 + cb;
#pragma unroll
    for (int i = 0; i < 32; i += 8) {
      float4 a = *(const float4*)(src + i);
      float4 b = *(const float4*)(src + i + 4);
      v8h h;
      h[0] = (_Float16)(a.x * scale); h[1] = (_Float16)(a.y * scale);
      h[2] = (_Float16)(a.z * scale); h[3] = (_Float16)(a.w * scale);
      h[4] = (_Float16)(b.x * scale); h[5] = (_Float16)(b.y * scale);
      h[6] = (_Float16)(b.z * scale); h[7] = (_Float16)(b.w * scale);
      *(v8h*)(dst + i) = h;
    }
  }
  __syncthreads();

  // Per-wave Q A-fragments (rows wave*16..+15), two DK slices — loop invariant.
  v16h aq0 = load_frag16(Qs, wave * 16, 64, 0, lane);
  v16h aq1 = load_frag16(Qs, wave * 16, 64, 32, lane);

  // All-ones B operand for the row-sum WMMA (P · 1 -> every column = rowsum).
  v16h onesB;
#pragma unroll
  for (int i = 0; i < 16; ++i) onesB[i] = (_Float16)1.0f;

  float m[8], l[8];
  v8f o0 = {}, o1 = {}, o2 = {}, o3 = {};
#pragma unroll
  for (int r = 0; r < 8; ++r) { m[r] = -__builtin_inff(); l[r] = 0.0f; }

  const int qw0  = q0 + wave * 16;
  const int qw0u = __builtin_amdgcn_readfirstlane(qw0);   // wave-uniform -> SGPR
  const int numKT = (q0 + 64) >> 5;     // causal: k-tiles of 32 up to q tile end

  for (int kt = 0; kt < numKT; ++kt) {
    const int kb = kt << 5;

    // ---- Cooperative stage: K row-major f16, V transposed f16 ----
    {
      const int krow = tid >> 2;              // 0..31
      const int cb   = (tid & 3) * 16;
      const float* ksrc = Kg + base + (size_t)(kb + krow) * ADK + cb;
      _Float16* kdst = Ks + krow * 64 + cb;
#pragma unroll
      for (int i = 0; i < 16; i += 8) {
        float4 a = *(const float4*)(ksrc + i);
        float4 b = *(const float4*)(ksrc + i + 4);
        v8h h;
        h[0] = (_Float16)a.x; h[1] = (_Float16)a.y;
        h[2] = (_Float16)a.z; h[3] = (_Float16)a.w;
        h[4] = (_Float16)b.x; h[5] = (_Float16)b.y;
        h[6] = (_Float16)b.z; h[7] = (_Float16)b.w;
        *(v8h*)(kdst + i) = h;
      }
      const float* vsrc = Vg + base + (size_t)(kb + krow) * ADV + cb;
#pragma unroll
      for (int i = 0; i < 16; i += 4) {
        float4 a = *(const float4*)(vsrc + i);
        VTs[(cb + i + 0) * 32 + krow] = (_Float16)a.x;
        VTs[(cb + i + 1) * 32 + krow] = (_Float16)a.y;
        VTs[(cb + i + 2) * 32 + krow] = (_Float16)a.z;
        VTs[(cb + i + 3) * 32 + krow] = (_Float16)a.w;
      }
      if (kt + 1 < numKT) {
        __builtin_prefetch(Kg + base + (size_t)(kb + 32 + krow) * ADK, 0, 0);
      }
    }
    __syncthreads();

    if (kb <= qw0u + 15) {   // scalar branch: skip fully-masked tiles
      // ---- S = Q·K^T (base-2 scaled) : batch B-fragment loads, then 4 WMMAs ----
      v16h b00 = load_frag16(Ks,  0, 64,  0, lane);
      v16h b10 = load_frag16(Ks, 16, 64,  0, lane);
      v16h b01 = load_frag16(Ks,  0, 64, 32, lane);
      v16h b11 = load_frag16(Ks, 16, 64, 32, lane);
      v8f s0 = {}, s1 = {};
      s0 = __builtin_amdgcn_wmma_f32_16x16x32_f16(false, aq0, false, b00, (short)0, s0, false, false);
      s1 = __builtin_amdgcn_wmma_f32_16x16x32_f16(false, aq0, false, b10, (short)0, s1, false, false);
      s0 = __builtin_amdgcn_wmma_f32_16x16x32_f16(false, aq1, false, b01, (short)0, s0, false, false);
      s1 = __builtin_amdgcn_wmma_f32_16x16x32_f16(false, aq1, false, b11, (short)0, s1, false, false);

      // ---- Online softmax (exp2 domain). Fast path for sub-diagonal tiles ----
      _Float16* pw = Ps + wave * (16 * 32);
      float alpha[8];
      if (kb + 31 <= qw0u) {          // scalar branch: no masking needed
#pragma unroll
        for (int r = 0; r < 8; ++r) {
          const float mx = rowmax16(fmaxf(s0[r], s1[r]));
          const float mn = fmaxf(m[r], mx);
          alpha[r] = __builtin_amdgcn_exp2f(m[r] - mn);
          const float p0 = __builtin_amdgcn_exp2f(s0[r] - mn);
          const float p1 = __builtin_amdgcn_exp2f(s1[r] - mn);
          m[r] = mn;
          o0[r] *= alpha[r]; o1[r] *= alpha[r]; o2[r] *= alpha[r]; o3[r] *= alpha[r];
          pw[(r + 8 * lh) * 32 + ln]      = (_Float16)p0;
          pw[(r + 8 * lh) * 32 + 16 + ln] = (_Float16)p1;
        }
      } else {                         // diagonal tile: apply causal mask
#pragma unroll
        for (int r = 0; r < 8; ++r) {
          const int grow = qw0 + r + 8 * lh;
          const float x0 = ((kb + ln)      <= grow) ? s0[r] : -__builtin_inff();
          const float x1 = ((kb + 16 + ln) <= grow) ? s1[r] : -__builtin_inff();
          const float mx = rowmax16(fmaxf(x0, x1));
          const float mn = fmaxf(m[r], mx);
          alpha[r] = __builtin_amdgcn_exp2f(m[r] - mn);  // 0 on first visit
          const float p0 = __builtin_amdgcn_exp2f(x0 - mn);
          const float p1 = __builtin_amdgcn_exp2f(x1 - mn);
          m[r] = mn;
          o0[r] *= alpha[r]; o1[r] *= alpha[r]; o2[r] *= alpha[r]; o3[r] *= alpha[r];
          pw[(r + 8 * lh) * 32 + ln]      = (_Float16)p0;
          pw[(r + 8 * lh) * 32 + 16 + ln] = (_Float16)p1;
        }
      }

      // ---- O += P·V and l += rowsum(P) via WMMA (batch loads first) ----
      v16h pa  = load_frag16(pw, 0, 32, 0, lane);   // same-wave DS is in-order
      v16h bv0 = load_frag16(VTs,  0, 32, 0, lane);
      v16h bv1 = load_frag16(VTs, 16, 32, 0, lane);
      v16h bv2 = load_frag16(VTs, 32, 32, 0, lane);
      v16h bv3 = load_frag16(VTs, 48, 32, 0, lane);
      v8f rs = {};
      rs = __builtin_amdgcn_wmma_f32_16x16x32_f16(false, pa, false, onesB, (short)0, rs, false, false);
      o0 = __builtin_amdgcn_wmma_f32_16x16x32_f16(false, pa, false, bv0, (short)0, o0, false, false);
      o1 = __builtin_amdgcn_wmma_f32_16x16x32_f16(false, pa, false, bv1, (short)0, o1, false, false);
      o2 = __builtin_amdgcn_wmma_f32_16x16x32_f16(false, pa, false, bv2, (short)0, o2, false, false);
      o3 = __builtin_amdgcn_wmma_f32_16x16x32_f16(false, pa, false, bv3, (short)0, o3, false, false);
#pragma unroll
      for (int r = 0; r < 8; ++r) l[r] = l[r] * alpha[r] + rs[r];
    }
    __syncthreads();
  }

  // ---- Epilogue: normalize by l (single v_rcp_f32) and store fp32 ----
  float* obase = Og + (size_t)bh * AS * ADV;
#pragma unroll
  for (int r = 0; r < 8; ++r) {
    const int grow = qw0 + r + 8 * lh;
    const float inv = __builtin_amdgcn_rcpf(l[r]);   // causal diag => l > 0
    float* op = obase + (size_t)grow * ADV + ln;
    op[0]  = o0[r] * inv;
    op[16] = o1[r] * inv;
    op[32] = o2[r] * inv;
    op[48] = o3[r] * inv;
  }
}

extern "C" void kernel_launch(void* const* d_in, const int* in_sizes, int n_in,
                              void* d_out, int out_size, void* d_ws, size_t ws_size,
                              hipStream_t stream) {
  const float* Q = (const float*)d_in[0];
  const float* K = (const float*)d_in[1];
  const float* V = (const float*)d_in[2];
  // d_in[3] is the causal mask; causality is applied analytically in-kernel.
  float* O = (float*)d_out;
  (void)in_sizes; (void)n_in; (void)out_size; (void)d_ws; (void)ws_size;

  dim3 grid(AS / 64, AB * AH);   // (32, 32)
  dim3 block(128);               // 4 wave32
  hipLaunchKernelGGL(fa_fwd_f16wmma, grid, block, 0, stream, Q, K, V, O);
}
